// HeteroGNN_89137751261399
// MI455X (gfx1250) — matrix operand order repaired and checked by reading, not compiled, not run on hardware
//
#include <hip/hip_runtime.h>
#include <hip/hip_bf16.h>

// ---------------------------------------------------------------------------
// HeteroGNN on MI455X (gfx1250, wave32).
// Memory-bound (~1.8 GB gather/scatter vs ~23 GFLOP GEMM). GEMMs run on
// v_wmma_f32_16x16x32_bf16 (bf16 operands, f32 accumulate); scatter uses
// hardware global_atomic_add_f32. Staging uses B128 global loads and
// packed B64 LDS stores.
// ---------------------------------------------------------------------------

typedef __attribute__((ext_vector_type(16))) __bf16 v16bf;
typedef __attribute__((ext_vector_type(8)))  __bf16 v8bf;
typedef __attribute__((ext_vector_type(8)))  float  v8f;

__device__ __forceinline__ unsigned f2bf_bits(float f) {
    union { float f; unsigned u; } in; in.f = f;
    unsigned r = in.u + 0x7FFFu + ((in.u >> 16) & 1u);   // round-to-nearest-even
    return r >> 16;
}

// pack 4 floats -> 4 bf16 in a uint2 (one ds_store_b64)
__device__ __forceinline__ uint2 pack4bf(float a, float b, float c, float d) {
    uint2 p;
    p.x = f2bf_bits(a) | (f2bf_bits(b) << 16);
    p.y = f2bf_bits(c) | (f2bf_bits(d) << 16);
    return p;
}

__device__ __forceinline__ void atomic_fadd(float* p, float v) {
    // Guaranteed single-instruction hardware float atomic (no CAS loop).
    asm volatile("global_atomic_add_f32 %0, %1, off" :: "v"(p), "v"(v) : "memory");
}

// ---------------------------------------------------------------------------
// Kernel 0: zero the scatter accumulators in workspace
// ---------------------------------------------------------------------------
__global__ void __launch_bounds__(256) hgnn_zero(float* __restrict__ p, long long n) {
    long long i = (long long)blockIdx.x * blockDim.x + threadIdx.x;
    if (i < n) p[i] = 0.0f;
}

// ---------------------------------------------------------------------------
// WMMA fragment helpers (shared by both GEMM kernels).
// lA: [128 rows][32 k] bf16 row-major; lB: [8 ntiles][16 n][32 k] bf16.
// ---------------------------------------------------------------------------
__device__ __forceinline__ v16bf load_a_frag(const __bf16* lA, int mrow, int half) {
    v8bf alo = *(const v8bf*)&lA[mrow * 32 + 8 * half];
    v8bf ahi = *(const v8bf*)&lA[mrow * 32 + 16 + 8 * half];
    v16bf a;
    #pragma unroll
    for (int i = 0; i < 8; ++i) { a[i] = alo[i]; a[8 + i] = ahi[i]; }
    return a;
}

// ---------------------------------------------------------------------------
// Kernel 1: u = x_user[M,256] @ W_user[256,128] + b_user   (f32 out)
// 256 threads = 8 waves; block covers 128 rows; wave owns 16 rows x 128 cols.
// ---------------------------------------------------------------------------
__global__ void __launch_bounds__(256)
hgnn_gemm_user(const float* __restrict__ X, const float* __restrict__ W,
               const float* __restrict__ bias, float* __restrict__ U, int M)
{
    __shared__ alignas(32) __bf16 lA[128 * 32];      // [row][k]
    __shared__ alignas(32) __bf16 lB[8 * 16 * 32];   // [ntile][nlocal][k]

    const int tid  = threadIdx.x;
    const int lane = tid & 31;
    const int wave = tid >> 5;
    const int row0 = blockIdx.x * 128;
    const int half = lane >> 4;
    const int l16  = lane & 15;
    const bool full = (row0 + 128 <= M);

    v8f acc[8] = {};

    for (int k0 = 0; k0 < 256; k0 += 32) {
        // stage A: 128x32 tile as float4 loads + packed b64 LDS stores
        #pragma unroll
        for (int i = 0; i < 4; ++i) {
            int idx = tid + i * 256;                 // 0..1023 float4 slots
            int r = idx >> 3, q = idx & 7;           // q = float4 within row
            int row = row0 + r; if (row >= M) row = M - 1;
            const float4 x = *(const float4*)&X[(long long)row * 256 + k0 + q * 4];
            *(uint2*)&lA[r * 32 + q * 4] = pack4bf(x.x, x.y, x.z, x.w);
        }
        // stage B transposed: thread owns (n, 4 consecutive k); coalesced per k
        #pragma unroll
        for (int i = 0; i < 4; ++i) {
            int idx = tid + i * 256;                 // 0..1023
            int kg = idx >> 7, n = idx & 127;        // kks = kg*4
            int kks = kg * 4;
            float w0 = W[(long long)(k0 + kks + 0) * 128 + n];
            float w1 = W[(long long)(k0 + kks + 1) * 128 + n];
            float w2 = W[(long long)(k0 + kks + 2) * 128 + n];
            float w3 = W[(long long)(k0 + kks + 3) * 128 + n];
            *(uint2*)&lB[(n >> 4) * 512 + (n & 15) * 32 + kks] = pack4bf(w0, w1, w2, w3);
        }
        __syncthreads();

        const v16bf a = load_a_frag(lA, wave * 16 + l16, half);
        #pragma unroll
        for (int nt = 0; nt < 8; ++nt) {
            v16bf b = load_a_frag(lB + nt * 512, l16, half);
            acc[nt] = __builtin_amdgcn_wmma_f32_16x16x32_bf16(
                false, a, false, b, (short)0, acc[nt], false, false);
        }
        __syncthreads();
    }

    // epilogue: + bias (uniform guard: only last block is partial)
    if (full) {
        #pragma unroll
        for (int nt = 0; nt < 8; ++nt) {
            int col = nt * 16 + l16;
            float bv = bias[col];
            #pragma unroll
            for (int r = 0; r < 8; ++r) {
                int row = row0 + wave * 16 + half * 8 + r;
                U[(long long)row * 128 + col] = acc[nt][r] + bv;
            }
        }
    } else {
        #pragma unroll
        for (int nt = 0; nt < 8; ++nt) {
            int col = nt * 16 + l16;
            float bv = bias[col];
            #pragma unroll
            for (int r = 0; r < 8; ++r) {
                int row = row0 + wave * 16 + half * 8 + r;
                if (row < M) U[(long long)row * 128 + col] = acc[nt][r] + bv;
            }
        }
    }
}

// ---------------------------------------------------------------------------
// Kernel 2: edge scatter-add. One wave per edge (lanes share e), edge indices
// forced onto the scalar path, float4 coalesced gather, 4x
// global_atomic_add_f32 into agg, lane 0 bumps the count.
// ---------------------------------------------------------------------------
__global__ void __launch_bounds__(256)
hgnn_scatter(const float* __restrict__ Xsrc, const int* __restrict__ src,
             const int* __restrict__ dst, float* __restrict__ agg,
             float* __restrict__ cnt, int E)
{
    long long t = (long long)blockIdx.x * blockDim.x + threadIdx.x;
    int e = (int)(t >> 5);          // wave-uniform: 32 consecutive tids share e
    int g = (int)(t & 31);
    if (e >= E) return;
    int ew = __builtin_amdgcn_readfirstlane(e);     // scalarize index loads
    int s  = src[ew];
    int d  = dst[ew];
    const float4 v = *(const float4*)&Xsrc[(long long)s * 128 + g * 4];
    float* p = &agg[(long long)d * 128 + g * 4];
    atomic_fadd(p + 0, v.x);
    atomic_fadd(p + 1, v.y);
    atomic_fadd(p + 2, v.z);
    atomic_fadd(p + 3, v.w);
    if (g == 0) atomic_fadd(&cnt[d], 1.0f);
}

// ---------------------------------------------------------------------------
// Kernel 3: fused finalize.
//   out = relu( mean_img@Wl_img + mean_txt@Wl_txt + u@(Wr_img+Wr_txt)
//               + bl_img + bl_txt )
// as one K=384 GEMM; mean = agg * (1/max(cnt,1)) computed during A staging.
// ---------------------------------------------------------------------------
__global__ void __launch_bounds__(256)
hgnn_fuse(const float* __restrict__ aggI, const float* __restrict__ cntI,
          const float* __restrict__ aggT, const float* __restrict__ cntT,
          const float* __restrict__ U,
          const float* __restrict__ WlI, const float* __restrict__ WlT,
          const float* __restrict__ WrI, const float* __restrict__ WrT,
          const float* __restrict__ blI, const float* __restrict__ blT,
          float* __restrict__ out, int M)
{
    __shared__ alignas(32) __bf16 lA[128 * 32];
    __shared__ alignas(32) __bf16 lB[8 * 16 * 32];

    const int tid  = threadIdx.x;
    const int lane = tid & 31;
    const int wave = tid >> 5;
    const int row0 = blockIdx.x * 128;
    const int half = lane >> 4;
    const int l16  = lane & 15;
    const bool full = (row0 + 128 <= M);

    v8f acc[8] = {};

    for (int k0 = 0; k0 < 384; k0 += 32) {
        const int blk   = k0 >> 7;     // 0: img-mean, 1: txt-mean, 2: u
        const int kbase = k0 & 127;

        // stage A (float4 loads, packed b64 LDS stores)
        #pragma unroll
        for (int i = 0; i < 4; ++i) {
            int idx = tid + i * 256;
            int r = idx >> 3, q = idx & 7;
            int row = row0 + r; if (row >= M) row = M - 1;
            int kc = kbase + q * 4;
            float4 x;
            if (blk == 0) {
                x = *(const float4*)&aggI[(long long)row * 128 + kc];
                float inv = 1.0f / fmaxf(cntI[row], 1.0f);
                x.x *= inv; x.y *= inv; x.z *= inv; x.w *= inv;
            } else if (blk == 1) {
                x = *(const float4*)&aggT[(long long)row * 128 + kc];
                float inv = 1.0f / fmaxf(cntT[row], 1.0f);
                x.x *= inv; x.y *= inv; x.z *= inv; x.w *= inv;
            } else {
                x = *(const float4*)&U[(long long)row * 128 + kc];
            }
            *(uint2*)&lA[r * 32 + q * 4] = pack4bf(x.x, x.y, x.z, x.w);
        }
        // stage B: [Wl_img ; Wl_txt ; Wr_img + Wr_txt], transposed per n-tile
        #pragma unroll
        for (int i = 0; i < 4; ++i) {
            int idx = tid + i * 256;
            int kg = idx >> 7, n = idx & 127;
            int kks = kg * 4;
            float w[4];
            #pragma unroll
            for (int j = 0; j < 4; ++j) {
                int kglob = k0 + kks + j;
                if (kglob < 128)      w[j] = WlI[(long long)kglob * 128 + n];
                else if (kglob < 256) w[j] = WlT[(long long)(kglob - 128) * 128 + n];
                else                  w[j] = WrI[(long long)(kglob - 256) * 128 + n] +
                                             WrT[(long long)(kglob - 256) * 128 + n];
            }
            *(uint2*)&lB[(n >> 4) * 512 + (n & 15) * 32 + kks] = pack4bf(w[0], w[1], w[2], w[3]);
        }
        __syncthreads();

        const v16bf a = load_a_frag(lA, wave * 16 + l16, half);
        #pragma unroll
        for (int nt = 0; nt < 8; ++nt) {
            v16bf b = load_a_frag(lB + nt * 512, l16, half);
            acc[nt] = __builtin_amdgcn_wmma_f32_16x16x32_bf16(
                false, a, false, b, (short)0, acc[nt], false, false);
        }
        __syncthreads();
    }

    // epilogue: summed biases + ReLU (uniform guard)
    if (full) {
        #pragma unroll
        for (int nt = 0; nt < 8; ++nt) {
            int col = nt * 16 + l16;
            float bv = blI[col] + blT[col];
            #pragma unroll
            for (int r = 0; r < 8; ++r) {
                int row = row0 + wave * 16 + half * 8 + r;
                out[(long long)row * 128 + col] = fmaxf(acc[nt][r] + bv, 0.0f);
            }
        }
    } else {
        #pragma unroll
        for (int nt = 0; nt < 8; ++nt) {
            int col = nt * 16 + l16;
            float bv = blI[col] + blT[col];
            #pragma unroll
            for (int r = 0; r < 8; ++r) {
                int row = row0 + wave * 16 + half * 8 + r;
                if (row < M)
                    out[(long long)row * 128 + col] = fmaxf(acc[nt][r] + bv, 0.0f);
            }
        }
    }
}

// ---------------------------------------------------------------------------
// Host-side launch
// ---------------------------------------------------------------------------
extern "C" void kernel_launch(void* const* d_in, const int* in_sizes, int n_in,
                              void* d_out, int out_size, void* d_ws, size_t ws_size,
                              hipStream_t stream)
{
    const float* x_user = (const float*)d_in[0];
    const float* x_img  = (const float*)d_in[1];
    const float* x_txt  = (const float*)d_in[2];
    const int*   e_img  = (const int*)d_in[3];   // [2, E]: row0 = src, row1 = dst
    const int*   e_txt  = (const int*)d_in[4];
    const float* W_user = (const float*)d_in[5];
    const float* b_user = (const float*)d_in[6];
    const float* Wl_img = (const float*)d_in[7];
    const float* bl_img = (const float*)d_in[8];
    const float* Wr_img = (const float*)d_in[9];
    const float* Wl_txt = (const float*)d_in[10];
    const float* bl_txt = (const float*)d_in[11];
    const float* Wr_txt = (const float*)d_in[12];

    const int M = in_sizes[0] / 256;      // 100000
    const int E = in_sizes[3] / 2;        // 800000

    // workspace layout (f32): u | agg_img | agg_txt | cnt_img | cnt_txt
    float* u    = (float*)d_ws;
    float* aggI = u    + (long long)M * 128;
    float* aggT = aggI + (long long)M * 128;
    float* cntI = aggT + (long long)M * 128;
    float* cntT = cntI + M;

    // 0) zero the atomic accumulators (agg_img .. cnt_txt)
    {
        long long nz = 2LL * M * 128 + 2LL * M;
        int blocks = (int)((nz + 255) / 256);
        hgnn_zero<<<blocks, 256, 0, stream>>>(aggI, nz);
    }

    // 1) u = x_user @ W_user + b_user
    {
        int blocks = (M + 127) / 128;
        hgnn_gemm_user<<<blocks, 256, 0, stream>>>(x_user, W_user, b_user, u, M);
    }

    // 2) scatter-add both relations
    {
        long long nthreads = (long long)E * 32;
        int blocks = (int)((nthreads + 255) / 256);
        hgnn_scatter<<<blocks, 256, 0, stream>>>(x_img, e_img, e_img + E, aggI, cntI, E);
        hgnn_scatter<<<blocks, 256, 0, stream>>>(x_txt, e_txt, e_txt + E, aggT, cntT, E);
    }

    // 3) fused mean + 3 GEMMs + bias + ReLU
    {
        int blocks = (M + 127) / 128;
        hgnn_fuse<<<blocks, 256, 0, stream>>>(aggI, cntI, aggT, cntT, u,
                                              Wl_img, Wl_txt, Wr_img, Wr_txt,
                                              bl_img, bl_txt,
                                              (float*)d_out, M);
    }
}
